// SwinTransformerBlock5D_26130581028953
// MI455X (gfx1250) — matrix operand order repaired and checked by reading, not compiled, not run on hardware
//
#include <hip/hip_runtime.h>

// ---------------------------------------------------------------------------
// Types for CDNA5 WMMA (wave32, 16x16x32 bf16 -> f32)
// ---------------------------------------------------------------------------
typedef __attribute__((ext_vector_type(16))) __bf16 v16bf;
typedef __attribute__((ext_vector_type(8)))  float  v8f;
typedef __attribute__((ext_vector_type(4)))  unsigned int v4u;
typedef __attribute__((ext_vector_type(4)))  int v4i;
typedef __attribute__((ext_vector_type(8)))  int v8i;

#define WMMA_BF16(a, b, c) \
    __builtin_amdgcn_wmma_f32_16x16x32_bf16(false, (a), false, (b), (short)0, (c), false, false)

// Problem constants (from the reference)
#define NW    512     // number of windows
#define NTOK  256     // tokens per window
#define CCH   128     // channels
#define HEADS 4
#define HD    32
#define TOTT  131072  // total tokens (1*16*16*8*8*8)

// ---------------------------------------------------------------------------
// Tensor Data Mover staging: copy rows x cols bf16 elements (contiguous,
// row stride == cols) from global memory into LDS.  D# fields per ISA
// CDNA5 ISA.  Issued by one wave; EXEC is ignored.
// ---------------------------------------------------------------------------
#if __has_builtin(__builtin_amdgcn_tensor_load_to_lds)
#define HAVE_TDM 1
#else
#define HAVE_TDM 0
#endif

#if HAVE_TDM
__device__ inline void tdm_load_2d(const void* lds_ptr, const void* gptr,
                                   unsigned cols, unsigned rows) {
    unsigned long long ga = (unsigned long long)gptr;
    // generic LDS pointer = aperture | lds_offset; low 32 bits are the LDS addr
    unsigned lds_off = (unsigned)(unsigned long long)lds_ptr;
    v4u g0;
    g0[0] = 1u;                                                  // count=1 (valid user D#)
    g0[1] = lds_off;                                             // lds_addr
    g0[2] = (unsigned)ga;                                        // global_addr[31:0]
    g0[3] = (unsigned)((ga >> 32) & 0x01ffffffu) | 0x80000000u;  // addr[56:32] | type=2
    v8i g1;
    g1[0] = (int)(1u << 16);                                     // wg_mask=0, data_size=1 (2B)
    g1[1] = (int)((cols & 0xffffu) << 16);                       // tensor_dim0[15:0]
    g1[2] = (int)(((cols >> 16) & 0xffffu) | ((rows & 0xffffu) << 16)); // dim0 hi | dim1 lo
    g1[3] = (int)(((rows >> 16) & 0xffffu) | ((cols & 0xffffu) << 16)); // dim1 hi | tile_dim0
    g1[4] = (int)(rows & 0xffffu);                               // tile_dim1 (tile_dim2=0)
    g1[5] = (int)cols;                                           // tensor_dim0_stride[31:0]
    g1[6] = 0;                                                   // stride hi | dim1_stride lo
    g1[7] = 0;
    v4i z4 = {0, 0, 0, 0};
#if __clang_major__ >= 23
    v8i z8 = {0, 0, 0, 0, 0, 0, 0, 0};
    __builtin_amdgcn_tensor_load_to_lds(g0, g1, z4, z4, z8, 0);
#else
    __builtin_amdgcn_tensor_load_to_lds(g0, g1, z4, z4, 0);
#endif
}

__device__ inline void tdm_wait() {
#if __has_builtin(__builtin_amdgcn_s_wait_tensorcnt)
    __builtin_amdgcn_s_wait_tensorcnt(0);
#else
    asm volatile("s_wait_tensorcnt 0x0" ::: "memory");
#endif
}
#endif

// ---------------------------------------------------------------------------
// Fragment loaders (contiguous k = 16*half + s in A and B)
// ---------------------------------------------------------------------------
__device__ inline v16bf load_a_frag(const __bf16* base, int ld, int lane) {
    int row  = lane & 15;
    int half = lane >> 4;
    v16bf a;
#pragma unroll
    for (int j = 0; j < 8; ++j) {
        int kb = 2 * j + half * 16;  // 0,2,...,14 (+16)
        a[2 * j]     = base[row * ld + kb];
        a[2 * j + 1] = base[row * ld + kb + 1];
    }
    return a;
}

// B tile where column n is contiguous in memory: B[k][n] = base[n*ld + k]
__device__ inline v16bf load_b_frag(const __bf16* base, int ld, int lane) {
    int col  = lane & 15;
    int half = lane >> 4;
    v16bf b;
#pragma unroll
    for (int j = 0; j < 8; ++j) {
        int k = 2 * j + half * 16;
        b[2 * j]     = base[col * ld + k];
        b[2 * j + 1] = base[col * ld + k + 1];
    }
    return b;
}

// B tile stored K-major: B[k][n] = base[k*ld + n]  (used for P @ V)
__device__ inline v16bf load_b_frag_kmajor(const __bf16* base, int ld, int lane) {
    int col  = lane & 15;
    int half = lane >> 4;
    v16bf b;
#pragma unroll
    for (int j = 0; j < 8; ++j) {
        int k = 2 * j + half * 16;
        b[2 * j]     = base[k * ld + col];
        b[2 * j + 1] = base[(k + 1) * ld + col];
    }
    return b;
}

// ---------------------------------------------------------------------------
// (window, token-in-window) -> source/destination token (bijective map)
// ---------------------------------------------------------------------------
__device__ inline int win_token_src(int wi, int n) {
    int iv = n & 1, iu = (n >> 1) & 1, iw = (n >> 2) & 3, ih = (n >> 4) & 3, id = (n >> 6) & 3;
    int v0 = wi & 3, u0 = (wi >> 2) & 3, w0 = (wi >> 4) & 1, h0 = (wi >> 5) & 3, d0 = (wi >> 7) & 3;
    int d = ((d0 << 2) + id + 2) & 15;
    int h = ((h0 << 2) + ih + 2) & 15;
    int w = ((w0 << 2) + iw + 2) & 7;
    int u = ((u0 << 1) + iu + 1) & 7;
    int v = ((v0 << 1) + iv + 1) & 7;
    return ((((d * 16 + h) * 8 + w) * 8 + u) * 8) + v;
}

// ---------------------------------------------------------------------------
// Kernel 0: fp32 -> bf16 weight conversion
// ---------------------------------------------------------------------------
__global__ __launch_bounds__(256) void k_cvt(const float* __restrict__ src,
                                             __bf16* __restrict__ dst, int n) {
    int i = blockIdx.x * 256 + threadIdx.x;
    if (i < n) dst[i] = (__bf16)src[i];
}

// ---------------------------------------------------------------------------
// Kernel 1: LN1 + cyclic shift + 5D window partition -> bf16 (nW, N, C)
// ---------------------------------------------------------------------------
__global__ __launch_bounds__(256) void k_ln1(const float* __restrict__ x,
                                             const float* __restrict__ nw,
                                             const float* __restrict__ nb,
                                             __bf16* __restrict__ xw) {
    int t  = blockIdx.x * 256 + threadIdx.x;  // windowed token id
    int wi = t >> 8, n = t & 255;
    int src = win_token_src(wi, n);
    const float4* xr = (const float4*)(x + (long)src * CCH);
    float s = 0.f, ss = 0.f;
#pragma unroll 8
    for (int i = 0; i < 32; ++i) {
        float4 v = xr[i];
        s  += v.x + v.y + v.z + v.w;
        ss += v.x * v.x + v.y * v.y + v.z * v.z + v.w * v.w;
    }
    float mu  = s * (1.f / 128.f);
    float var = ss * (1.f / 128.f) - mu * mu;
    float inv = rsqrtf(var + 1e-5f);
    __bf16* o = xw + (long)t * CCH;
#pragma unroll 8
    for (int i = 0; i < 32; ++i) {
        float4 v = xr[i];
        o[4 * i + 0] = (__bf16)((v.x - mu) * inv * nw[4 * i + 0] + nb[4 * i + 0]);
        o[4 * i + 1] = (__bf16)((v.y - mu) * inv * nw[4 * i + 1] + nb[4 * i + 1]);
        o[4 * i + 2] = (__bf16)((v.z - mu) * inv * nw[4 * i + 2] + nb[4 * i + 2]);
        o[4 * i + 3] = (__bf16)((v.w - mu) * inv * nw[4 * i + 3] + nb[4 * i + 3]);
    }
}

// ---------------------------------------------------------------------------
// Kernel 2: QKV GEMM.  One block = half a window (128 tokens).
// ---------------------------------------------------------------------------
__global__ __launch_bounds__(256) void k_qkv(const __bf16* __restrict__ xw,
                                             const __bf16* __restrict__ wq,
                                             const float* __restrict__ qkv_b,
                                             __bf16* __restrict__ Q,
                                             __bf16* __restrict__ K,
                                             __bf16* __restrict__ V) {
    __shared__ __bf16 sX[128 * 128];  // 32 KB
    __shared__ __bf16 sW[384 * 128];  // 96 KB
    int bid = blockIdx.x, wi = bid >> 1, nb = bid & 1, tid = threadIdx.x;
    const __bf16* gx = xw + ((long)wi * 256 + nb * 128) * CCH;

#if HAVE_TDM
    if (tid == 0) {
        tdm_load_2d(sX, gx, 8192, 2);   // 16384 bf16
        tdm_load_2d(sW, wq, 8192, 6);   // 49152 bf16
        tdm_wait();
    }
    __syncthreads();
#else
    {
        const unsigned int* g = (const unsigned int*)gx;
        unsigned int* s = (unsigned int*)sX;
        for (int i = tid; i < 128 * 128 / 2; i += 256) s[i] = g[i];
        const unsigned int* g2 = (const unsigned int*)wq;
        unsigned int* s2 = (unsigned int*)sW;
        for (int i = tid; i < 384 * 128 / 2; i += 256) s2[i] = g2[i];
    }
    __syncthreads();
#endif

    int lane = tid & 31, wv = tid >> 5;
    int col = lane & 15, half = lane >> 4;
    for (int t = wv; t < 192; t += 8) {            // 8 m-tiles x 24 r-tiles
        int mt = t / 24, rt = t % 24;
        v8f acc = {};
#pragma unroll
        for (int kk = 0; kk < 4; ++kk) {
            v16bf a = load_a_frag(sX + mt * 16 * 128 + kk * 32, 128, lane);
            v16bf b = load_b_frag(sW + rt * 16 * 128 + kk * 32, 128, lane);
            acc = WMMA_BF16(a, b, acc);
        }
        int r = rt * 16 + col;
        int c = r & 127, hd = c >> 5, ch = c & 31;
        float bias  = qkv_b[r];
        float scale = (r < 128) ? 0.1767766952966369f : 1.0f;
        __bf16* dst = (r < 128) ? Q : ((r < 256) ? K : V);
#pragma unroll
        for (int i = 0; i < 8; ++i) {
            int n = nb * 128 + mt * 16 + i + half * 8;
            dst[(((long)wi * HEADS + hd) * NTOK + n) * HD + ch] = (__bf16)((acc[i] + bias) * scale);
        }
    }
}

// ---------------------------------------------------------------------------
// Kernel 3: windowed attention.  One block = (window, head).
// ---------------------------------------------------------------------------
__global__ __launch_bounds__(256) void k_attn(const __bf16* __restrict__ Q,
                                              const __bf16* __restrict__ K,
                                              const __bf16* __restrict__ V,
                                              const float* __restrict__ mask,
                                              __bf16* __restrict__ O) {
    __shared__ __bf16 sQ[NTOK * HD];     // 16 KB
    __shared__ __bf16 sK[NTOK * HD];     // 16 KB
    __shared__ __bf16 sV[NTOK * HD];     // 16 KB
    __shared__ float  sS[64 * NTOK];     // 64 KB
    __shared__ __bf16 sP[64 * NTOK];     // 32 KB
    __shared__ float  sRed[256];
    __shared__ float  sRowM[64];
    __shared__ float  sRowS[64];

    int b = blockIdx.x, wi = b >> 2, h = b & 3, tid = threadIdx.x;
    long hbase = ((long)wi * HEADS + h) * NTOK * HD;

#if HAVE_TDM
    if (tid == 0) {
        tdm_load_2d(sQ, Q + hbase, 8192, 1);
        tdm_load_2d(sK, K + hbase, 8192, 1);
        tdm_load_2d(sV, V + hbase, 8192, 1);
        tdm_wait();
    }
    __syncthreads();
#else
    {
        const unsigned int* gq = (const unsigned int*)(Q + hbase);
        const unsigned int* gk = (const unsigned int*)(K + hbase);
        const unsigned int* gv = (const unsigned int*)(V + hbase);
        unsigned int* q = (unsigned int*)sQ;
        unsigned int* k = (unsigned int*)sK;
        unsigned int* v = (unsigned int*)sV;
        for (int i = tid; i < NTOK * HD / 2; i += 256) { q[i] = gq[i]; k[i] = gk[i]; v[i] = gv[i]; }
    }
    __syncthreads();
#endif

    int lane = tid & 31, wv = tid >> 5;
    int col = lane & 15, half = lane >> 4;
    const float* mbase = mask + (long)wi * NTOK * NTOK;
    // prefetch the first mask row-block (each thread touches a distinct line)
    __builtin_prefetch((const void*)(mbase + tid * 64), 0, 0);

    for (int rb = 0; rb < 4; ++rb) {
        if (rb < 3)  // prefetch next 64-row mask block (64 KB) while computing
            __builtin_prefetch((const void*)(mbase + (long)(rb + 1) * 64 * NTOK + tid * 64), 0, 0);

        // ---- S tile block: 64 rows x 256 cols = 4 x 16 tiles, 8 per wave ----
        for (int t = wv; t < 64; t += 8) {
            int mt = t >> 4, nt = t & 15;
            v16bf a  = load_a_frag(sQ + (rb * 64 + mt * 16) * HD, HD, lane);
            v16bf bf = load_b_frag(sK + nt * 16 * HD, HD, lane);
            v8f acc = {};
            acc = WMMA_BF16(a, bf, acc);
#pragma unroll
            for (int i = 0; i < 8; ++i) {
                int rl = mt * 16 + i + half * 8;      // row within 64-block
                int rg = rb * 64 + rl;                // query token
                int cg = nt * 16 + col;               // key token
                sS[rl * NTOK + cg] = acc[i] + mbase[(long)rg * NTOK + cg];
            }
        }
        __syncthreads();

        // ---- softmax over 64 rows (4 segments of 64 columns per row) ----
        int r = tid & 63, seg = tid >> 6;
        const int s0 = seg * 64;
        float mx = -3.0e38f;
        for (int j = s0; j < s0 + 64; ++j) mx = fmaxf(mx, sS[r * NTOK + j]);
        sRed[seg * 64 + r] = mx;
        __syncthreads();
        if (tid < 64) {
            float m0 = fmaxf(fmaxf(sRed[tid], sRed[64 + tid]),
                             fmaxf(sRed[128 + tid], sRed[192 + tid]));
            sRowM[tid] = m0;
        }
        __syncthreads();
        float rm = sRowM[r], ps = 0.f;
        for (int j = s0; j < s0 + 64; ++j) {
            float e = __expf(sS[r * NTOK + j] - rm);
            sS[r * NTOK + j] = e;
            ps += e;
        }
        sRed[seg * 64 + r] = ps;
        __syncthreads();
        if (tid < 64)
            sRowS[tid] = sRed[tid] + sRed[64 + tid] + sRed[128 + tid] + sRed[192 + tid];
        __syncthreads();
        float inv = 1.0f / sRowS[r];
        for (int j = s0; j < s0 + 64; ++j)
            sP[r * NTOK + j] = (__bf16)(sS[r * NTOK + j] * inv);
        __syncthreads();

        // ---- O block: (64 x 256) @ (256 x 32) -> 4 x 2 tiles, 1 per wave ----
        {
            int mt = wv >> 1, nt = wv & 1;
            v8f acc = {};
#pragma unroll
            for (int kk = 0; kk < 8; ++kk) {
                v16bf a  = load_a_frag(sP + mt * 16 * NTOK + kk * 32, NTOK, lane);
                v16bf bf = load_b_frag_kmajor(sV + kk * 32 * HD + nt * 16, HD, lane);
                acc = WMMA_BF16(a, bf, acc);
            }
#pragma unroll
            for (int i = 0; i < 8; ++i) {
                int n = rb * 64 + mt * 16 + i + half * 8;
                int c = h * HD + nt * 16 + col;
                O[((long)wi * NTOK + n) * CCH + c] = (__bf16)acc[i];
            }
        }
        __syncthreads();
    }
}

// ---------------------------------------------------------------------------
// Kernel 4: proj GEMM + reverse partition/shift + residual -> d_out (= x2)
// ---------------------------------------------------------------------------
__global__ __launch_bounds__(256) void k_proj(const __bf16* __restrict__ O,
                                              const __bf16* __restrict__ pw,
                                              const float* __restrict__ pb,
                                              const float* __restrict__ x,
                                              float* __restrict__ out) {
    __shared__ __bf16 sO[NTOK * CCH];  // 64 KB
    __shared__ __bf16 sW[CCH * CCH];   // 32 KB
    int wi = blockIdx.x, tid = threadIdx.x;

#if HAVE_TDM
    if (tid == 0) {
        tdm_load_2d(sO, O + (long)wi * NTOK * CCH, 8192, 4);
        tdm_load_2d(sW, pw, 8192, 2);
        tdm_wait();
    }
    __syncthreads();
#else
    {
        const unsigned int* go = (const unsigned int*)(O + (long)wi * NTOK * CCH);
        unsigned int* so = (unsigned int*)sO;
        for (int i = tid; i < NTOK * CCH / 2; i += 256) so[i] = go[i];
        const unsigned int* gw = (const unsigned int*)pw;
        unsigned int* sw = (unsigned int*)sW;
        for (int i = tid; i < CCH * CCH / 2; i += 256) sw[i] = gw[i];
    }
    __syncthreads();
#endif

    int lane = tid & 31, wv = tid >> 5;
    int col = lane & 15, half = lane >> 4;
    for (int t = wv; t < 128; t += 8) {            // 16 m-tiles x 8 c-tiles
        int mt = t >> 3, ct = t & 7;
        v8f acc = {};
#pragma unroll
        for (int kk = 0; kk < 4; ++kk) {
            v16bf a = load_a_frag(sO + mt * 16 * CCH + kk * 32, CCH, lane);
            v16bf b = load_b_frag(sW + ct * 16 * CCH + kk * 32, CCH, lane);
            acc = WMMA_BF16(a, b, acc);
        }
        int c = ct * 16 + col;
        float bias = pb[c];
#pragma unroll
        for (int i = 0; i < 8; ++i) {
            int n   = mt * 16 + i + half * 8;
            int tok = win_token_src(wi, n);
            out[(long)tok * CCH + c] = x[(long)tok * CCH + c] + acc[i] + bias;
        }
    }
}

// ---------------------------------------------------------------------------
// Kernel 5: fused FFN.  One block = 16 tokens.  sW reused fc1 -> fc2.
// ---------------------------------------------------------------------------
__global__ __launch_bounds__(256) void k_ffn(const float* __restrict__ n2w,
                                             const float* __restrict__ n2b,
                                             const __bf16* __restrict__ w1,
                                             const float* __restrict__ b1,
                                             const __bf16* __restrict__ w2,
                                             const float* __restrict__ b2,
                                             float* __restrict__ out) {
    __shared__ __bf16 sW[512 * 128];   // 128 KB, reused fc1 then fc2
    __shared__ __bf16 sXn[16 * 128];   // 4 KB
    __shared__ __bf16 sH[16 * 512];    // 16 KB
    int tb = blockIdx.x * 16, tid = threadIdx.x;

#if HAVE_TDM
    if (tid == 0) tdm_load_2d(sW, w1, 8192, 8);   // 65536 bf16
#else
    {
        const unsigned int* g = (const unsigned int*)w1;
        unsigned int* s = (unsigned int*)sW;
        for (int i = tid; i < 512 * 128 / 2; i += 256) s[i] = g[i];
    }
#endif
    if (tid < 16) {                    // LN2 on this block's 16 tokens
        const float* row = out + (long)(tb + tid) * CCH;
        float s = 0.f, ss = 0.f;
        for (int i = 0; i < CCH; ++i) { float v = row[i]; s += v; ss += v * v; }
        float mu  = s * (1.f / 128.f);
        float inv = rsqrtf(ss * (1.f / 128.f) - mu * mu + 1e-5f);
        for (int i = 0; i < CCH; ++i)
            sXn[tid * CCH + i] = (__bf16)((row[i] - mu) * inv * n2w[i] + n2b[i]);
    }
#if HAVE_TDM
    if (tid == 0) tdm_wait();
#endif
    __syncthreads();

    int lane = tid & 31, wv = tid >> 5;
    int col = lane & 15, half = lane >> 4;

    // GEMM1: (16 x 128) @ (128 x 512) -> 32 n-tiles, 4 per wave, then GELU
    for (int t = wv; t < 32; t += 8) {
        v8f acc = {};
#pragma unroll
        for (int kk = 0; kk < 4; ++kk) {
            v16bf a = load_a_frag(sXn + kk * 32, CCH, lane);
            v16bf b = load_b_frag(sW + t * 16 * CCH + kk * 32, CCH, lane);
            acc = WMMA_BF16(a, b, acc);
        }
        int c = t * 16 + col;
        float bias = b1[c];
#pragma unroll
        for (int i = 0; i < 8; ++i) {
            float v = acc[i] + bias;
            v = 0.5f * v * (1.0f + erff(v * 0.7071067811865475f));  // exact GELU
            sH[(i + half * 8) * 512 + c] = (__bf16)v;
        }
    }
    __syncthreads();

#if HAVE_TDM
    if (tid == 0) { tdm_load_2d(sW, w2, 8192, 8); tdm_wait(); }
    __syncthreads();
#else
    {
        const unsigned int* g = (const unsigned int*)w2;
        unsigned int* s = (unsigned int*)sW;
        for (int i = tid; i < 128 * 512 / 2; i += 256) s[i] = g[i];
    }
    __syncthreads();
#endif

    // GEMM2: (16 x 512) @ (512 x 128) -> 8 n-tiles, 1 per wave
    {
        int nt = wv;
        v8f acc = {};
#pragma unroll
        for (int kk = 0; kk < 16; ++kk) {
            v16bf a = load_a_frag(sH + kk * 32, 512, lane);
            v16bf b = load_b_frag(sW + nt * 16 * 512 + kk * 32, 512, lane);
            acc = WMMA_BF16(a, b, acc);
        }
        int c = nt * 16 + col;
        float bias = b2[c];
#pragma unroll
        for (int i = 0; i < 8; ++i) {
            long tok = tb + i + half * 8;
            out[tok * CCH + c] += acc[i] + bias;
        }
    }
}

// ---------------------------------------------------------------------------
// Host launcher
// ---------------------------------------------------------------------------
extern "C" void kernel_launch(void* const* d_in, const int* in_sizes, int n_in,
                              void* d_out, int out_size, void* d_ws, size_t ws_size,
                              hipStream_t stream) {
    const float* x      = (const float*)d_in[0];
    const float* mask   = (const float*)d_in[1];
    const float* qkv_w  = (const float*)d_in[2];
    const float* qkv_b  = (const float*)d_in[3];
    const float* proj_w = (const float*)d_in[4];
    const float* proj_b = (const float*)d_in[5];
    const float* n1w    = (const float*)d_in[6];
    const float* n1b    = (const float*)d_in[7];
    const float* n2w    = (const float*)d_in[8];
    const float* n2b    = (const float*)d_in[9];
    const float* fc1w   = (const float*)d_in[10];
    const float* fc1b   = (const float*)d_in[11];
    const float* fc2w   = (const float*)d_in[12];
    const float* fc2b   = (const float*)d_in[13];
    float* out = (float*)d_out;
    char*  ws  = (char*)d_ws;

    // workspace layout (bytes)
    __bf16* qkvw_bf  = (__bf16*)(ws + 0);                //  96 KB (384x128)
    __bf16* projw_bf = (__bf16*)(ws + 98304);            //  32 KB (128x128)
    __bf16* fc1w_bf  = (__bf16*)(ws + 131072);           // 128 KB (512x128)
    __bf16* fc2w_bf  = (__bf16*)(ws + 262144);           // 128 KB (128x512)
    const size_t ACT = (size_t)NW * NTOK * CCH * 2;      // 33,554,432 B
    __bf16* xw = (__bf16*)(ws + 393216);                 // windowed LN'd x (bf16)
    __bf16* Ob = xw;                                     // reuse after k_qkv
    __bf16* Qb = (__bf16*)(ws + 393216 + ACT);
    __bf16* Kb = (__bf16*)(ws + 393216 + 2 * ACT);
    __bf16* Vb = (__bf16*)(ws + 393216 + 3 * ACT);

    k_cvt<<<(49152 + 255) / 256, 256, 0, stream>>>(qkv_w, qkvw_bf, 49152);
    k_cvt<<<(16384 + 255) / 256, 256, 0, stream>>>(proj_w, projw_bf, 16384);
    k_cvt<<<(65536 + 255) / 256, 256, 0, stream>>>(fc1w, fc1w_bf, 65536);
    k_cvt<<<(65536 + 255) / 256, 256, 0, stream>>>(fc2w, fc2w_bf, 65536);

    k_ln1<<<TOTT / 256, 256, 0, stream>>>(x, n1w, n1b, xw);
    k_qkv<<<NW * 2, 256, 0, stream>>>(xw, qkvw_bf, qkv_b, Qb, Kb, Vb);
    k_attn<<<NW * HEADS, 256, 0, stream>>>(Qb, Kb, Vb, mask, Ob);
    k_proj<<<NW, 256, 0, stream>>>(Ob, projw_bf, proj_b, x, out);
    k_ffn<<<TOTT / 16, 256, 0, stream>>>(n2w, n2b, fc1w_bf, fc1b, fc2w_bf, fc2b, out);
}
